// mLSTMCell_37048387895688
// MI455X (gfx1250) — matrix operand order repaired and checked by reading, not compiled, not run on hardware
//
#include <hip/hip_runtime.h>
#include <hip/hip_bf16.h>
#include <math.h>

// mLSTM parallel forward for MI455X (gfx1250, wave32, WMMA)
// B=2, S=2048, E=1024, NH=8, DH=128

#define B_  2
#define S_  2048
#define E_  1024
#define NH_ 8
#define DH_ 128
#define KV_TILE 32
#define ROWS 64                  // query rows per block (4 waves x 16)
#define NWAVES (ROWS / 16)
#define NTHR (NWAVES * 32)

#define QK_PAD 8                 // row pad (halves): 272B rows -> bank stride 4
#define VT_PAD 8                 // 40-half rows: 80B -> bank stride 20

typedef __attribute__((ext_vector_type(16))) _Float16 v16h;
typedef __attribute__((ext_vector_type(8)))  _Float16 v8h;
typedef __attribute__((ext_vector_type(4)))  _Float16 v4h;
typedef __attribute__((ext_vector_type(8)))  float    v8f;
typedef __attribute__((ext_vector_type(4)))  float    v4f;

__device__ __forceinline__ float hsum16(float x) {
#pragma unroll
  for (int m = 1; m < 16; m <<= 1) x += __shfl_xor(x, m, 32);
  return x;
}
__device__ __forceinline__ float logsig(float x) {
  return (x >= 0.f) ? -log1pf(expf(-x)) : (x - log1pf(expf(x)));
}
__device__ __forceinline__ v16h cat8(v8h lo, v8h hv) {
  return __builtin_shufflevector(lo, hv, 0, 1, 2, 3, 4, 5, 6, 7,
                                 8, 9, 10, 11, 12, 13, 14, 15);
}

// ---------------------------------------------------------------------------
// Kernel 1: gate projections  i_pre/f_pre[b,h,s] = [q|k|v] . w[h] + b[h]
// ---------------------------------------------------------------------------
__global__ __launch_bounds__(128) void mlstm_gates_kernel(
    const float* __restrict__ q, const float* __restrict__ k,
    const float* __restrict__ v,
    const float* __restrict__ igw, const float* __restrict__ igb,
    const float* __restrict__ fgw, const float* __restrict__ fgb,
    float* __restrict__ ipre, float* __restrict__ fpre)
{
  const int tok = blockIdx.x;          // b*S + s
  const int b   = tok / S_;
  const int s   = tok % S_;
  const int tid = threadIdx.x;
  const float* qb = q + (size_t)tok * E_;
  const float* kb = k + (size_t)tok * E_;
  const float* vb = v + (size_t)tok * E_;

  float acc[2 * NH_];
#pragma unroll
  for (int g = 0; g < 2 * NH_; ++g) acc[g] = 0.f;

  for (int idx = tid; idx < 3 * E_; idx += 128) {
    float x = (idx < E_) ? qb[idx] : (idx < 2 * E_) ? kb[idx - E_] : vb[idx - 2 * E_];
#pragma unroll
    for (int hh = 0; hh < NH_; ++hh) {
      acc[hh]       += x * igw[hh * 3 * E_ + idx];
      acc[NH_ + hh] += x * fgw[hh * 3 * E_ + idx];
    }
  }

  __shared__ float red[2 * NH_][128];
#pragma unroll
  for (int g = 0; g < 2 * NH_; ++g) red[g][tid] = acc[g];
  __syncthreads();
  for (int off = 64; off > 0; off >>= 1) {
    if (tid < off) {
#pragma unroll
      for (int g = 0; g < 2 * NH_; ++g) red[g][tid] += red[g][tid + off];
    }
    __syncthreads();
  }
  if (tid < NH_) {
    ipre[((size_t)b * NH_ + tid) * S_ + s] = red[tid][0] + igb[tid];
    fpre[((size_t)b * NH_ + tid) * S_ + s] = red[NH_ + tid][0] + fgb[tid];
  }
}

// ---------------------------------------------------------------------------
// Kernel 2: cs = cumsum(logsig(f)); g = i_pre - cs; M = prefix-max(g)
// ---------------------------------------------------------------------------
__global__ __launch_bounds__(32) void mlstm_scan_kernel(
    const float* __restrict__ fpre, const float* __restrict__ ipre,
    float* __restrict__ cs, float* __restrict__ Mv, float* __restrict__ Gv)
{
  const int bh   = blockIdx.x;
  const int lane = threadIdx.x;
  const float* f  = fpre + (size_t)bh * S_;
  const float* ip = ipre + (size_t)bh * S_;
  float* c = cs + (size_t)bh * S_;
  float* M = Mv + (size_t)bh * S_;
  float* G = Gv + (size_t)bh * S_;
  float carry = 0.f, carrym = -INFINITY;
  for (int base = 0; base < S_; base += 32) {
    float x = logsig(f[base + lane]);
#pragma unroll
    for (int d = 1; d < 32; d <<= 1) {
      float t = __shfl_up(x, d, 32);
      if (lane >= d) x += t;
    }
    x += carry;
    c[base + lane] = x;
    carry = __shfl(x, 31, 32);

    float gi = ip[base + lane] - x;
    G[base + lane] = gi;
#pragma unroll
    for (int d = 1; d < 32; d <<= 1) {
      float t = __shfl_up(gi, d, 32);
      if (lane >= d) gi = fmaxf(gi, t);
    }
    gi = fmaxf(gi, carrym);
    M[base + lane] = gi;
    carrym = __shfl(gi, 31, 32);
  }
}

// ---------------------------------------------------------------------------
// Kernel 3: streaming mLSTM attention + group norm.
// grid = B*NH*(S/ROWS) = 512 blocks; block = 128 threads (4 wave32).
// Stabilizer m_i = cs[i]+M[i] known a priori; w = exp(g[j]-M[i]).
// Row sums via one ones-matrix WMMA. All WMMA operands are contiguous
// b128 LDS loads (V staged transposed; rows padded for bank spread).
// ---------------------------------------------------------------------------
__global__ __launch_bounds__(NTHR) void mlstm_attn_kernel(
    const float* __restrict__ q, const float* __restrict__ k,
    const float* __restrict__ v,
    const float* __restrict__ Gv, const float* __restrict__ Mv,
    const float* __restrict__ cs,
    const float* __restrict__ nw, float* __restrict__ out)
{
  __shared__ _Float16 sQ[ROWS][DH_ + QK_PAD];          // 17 KB
  __shared__ _Float16 sK[KV_TILE][DH_ + QK_PAD];       // 8.5 KB
  __shared__ _Float16 sVt[DH_][KV_TILE + VT_PAD];      // 10 KB (V transposed)
  __shared__ _Float16 sP[NWAVES][16][KV_TILE + VT_PAD];// 5 KB
  __shared__ float sCsq[ROWS];
  __shared__ float sMq[ROWS];
  __shared__ float sG[KV_TILE];

  const int nblk = S_ / ROWS;
  const int ib = blockIdx.x % nblk;
  const int bh = blockIdx.x / nblk;
  const int b  = bh / NH_;
  const int h  = bh % NH_;
  const int r0 = ib * ROWS;

  const int tid  = threadIdx.x;
  const int wave = tid >> 5;
  const int lane = tid & 31;
  const int ln16 = lane & 15;
  const int hi   = lane >> 4;

  const float qscale = 0.08838834764831845f; // 1/sqrt(DH)
  const size_t bhS = (size_t)bh * S_;

  // ---- stage Q (pre-scaled, f32->f16, float4) and per-row cs/M
  for (int j = tid; j < ROWS * (DH_ / 4); j += NTHR) {
    int rr = j >> 5, dc = (j & 31) << 2;
    v4f val = *(const v4f*)(q + ((size_t)b * S_ + (r0 + rr)) * E_ + h * DH_ + dc);
    v4h hv;
    hv[0] = (_Float16)(val[0] * qscale); hv[1] = (_Float16)(val[1] * qscale);
    hv[2] = (_Float16)(val[2] * qscale); hv[3] = (_Float16)(val[3] * qscale);
    *(v4h*)&sQ[rr][dc] = hv;
  }
  if (tid < ROWS) {
    sCsq[tid] = cs[bhS + r0 + tid];
    sMq[tid]  = Mv[bhS + r0 + tid];
  }
  __syncthreads();

  // ---- Q in WMMA A-layout: two contiguous 8-half runs per chunk
  const int wr0 = wave * 16;
  v16h aQ[4];
#pragma unroll
  for (int c = 0; c < 4; ++c) {
    v8h lo = *(const v8h*)&sQ[wr0 + ln16][32 * c + 16 * hi];
    v8h hv = *(const v8h*)&sQ[wr0 + ln16][32 * c + 16 * hi + 8];
    aQ[c] = cat8(lo, hv);
  }
  float Mi[8];
#pragma unroll
  for (int vv = 0; vv < 8; ++vv) Mi[vv] = sMq[wr0 + vv + 8 * hi];

  v16h bOnes;
#pragma unroll
  for (int e = 0; e < 16; ++e) bOnes[e] = (_Float16)1.0f;

  v8f acc[8];
#pragma unroll
  for (int t = 0; t < 8; ++t) acc[t] = (v8f){0, 0, 0, 0, 0, 0, 0, 0};
  v8f sumAcc = (v8f){0, 0, 0, 0, 0, 0, 0, 0};

  const int nk = r0 + ROWS;
  for (int k0 = 0; k0 < nk; k0 += KV_TILE) {
    __syncthreads();
    if (k0 + KV_TILE < nk) {   // gfx1250 global_prefetch of next tile
      int rr = tid >> 2, dc = (tid & 3) << 5;
      __builtin_prefetch(k + ((size_t)b * S_ + (k0 + KV_TILE + rr)) * E_ + h * DH_ + dc, 0, 1);
      __builtin_prefetch(v + ((size_t)b * S_ + (k0 + KV_TILE + rr)) * E_ + h * DH_ + dc, 0, 1);
    }
    for (int j = tid; j < KV_TILE * (DH_ / 4); j += NTHR) {
      int rr = j >> 5, dc = (j & 31) << 2;
      size_t base = ((size_t)b * S_ + (k0 + rr)) * E_ + h * DH_ + dc;
      v4f kv  = *(const v4f*)(k + base);
      v4f vv4 = *(const v4f*)(v + base);
      v4h hk;
      hk[0] = (_Float16)kv[0]; hk[1] = (_Float16)kv[1];
      hk[2] = (_Float16)kv[2]; hk[3] = (_Float16)kv[3];
      *(v4h*)&sK[rr][dc] = hk;
      // V transposed: sVt[d][key]
      sVt[dc + 0][rr] = (_Float16)vv4[0];
      sVt[dc + 1][rr] = (_Float16)vv4[1];
      sVt[dc + 2][rr] = (_Float16)vv4[2];
      sVt[dc + 3][rr] = (_Float16)vv4[3];
    }
    if (tid < KV_TILE) sG[tid] = Gv[bhS + k0 + tid];
    __syncthreads();

    // ---- QK^T: two 16x16 f32 tiles, K=128 in 4 chunks (contiguous b128s)
    v8f qk0 = (v8f){0, 0, 0, 0, 0, 0, 0, 0};
    v8f qk1 = (v8f){0, 0, 0, 0, 0, 0, 0, 0};
#pragma unroll
    for (int c = 0; c < 4; ++c) {
      v16h bK0 = *(const v16h*)&sK[ln16][32 * c + 16 * hi];
      v16h bK1 = *(const v16h*)&sK[16 + ln16][32 * c + 16 * hi];
      qk0 = __builtin_amdgcn_wmma_f32_16x16x32_f16(false, aQ[c], false, bK0,
                                                   (short)0, qk0, false, false);
      qk1 = __builtin_amdgcn_wmma_f32_16x16x32_f16(false, aQ[c], false, bK1,
                                                   (short)0, qk1, false, false);
    }

    // ---- gating: w = exp(g[j] - M_i) (<=1), causal mask
    const float g0 = sG[ln16];
    const float g1 = sG[16 + ln16];
    const int colg0 = k0 + ln16;
    const int colg1 = k0 + 16 + ln16;
#pragma unroll
    for (int vv = 0; vv < 8; ++vv) {
      const int rowg = r0 + wr0 + vv + 8 * hi;
      float w0 = (colg0 <= rowg) ? expf(g0 - Mi[vv]) : 0.f;
      float w1 = (colg1 <= rowg) ? expf(g1 - Mi[vv]) : 0.f;
      float p0 = qk0[vv] * w0;
      float p1 = qk1[vv] * w1;
      sP[wave][vv + 8 * hi][ln16]      = (_Float16)p0;
      sP[wave][vv + 8 * hi][16 + ln16] = (_Float16)p1;
    }

    // wave-local LDS RAW before re-read
    __asm__ volatile("s_wait_dscnt 0x0" ::: "memory");

    // ---- P in A-layout (two contiguous 8-half runs)
    v16h aP;
    {
      v8h lo = *(const v8h*)&sP[wave][ln16][16 * hi];
      v8h hv = *(const v8h*)&sP[wave][ln16][16 * hi + 8];
      aP = cat8(lo, hv);
    }
    // row sums of P via ones-matrix WMMA
    sumAcc = __builtin_amdgcn_wmma_f32_16x16x32_f16(false, aP, false, bOnes,
                                                    (short)0, sumAcc, false, false);
    // ---- P(16x32) @ V(32x128): bV from transposed V = contiguous b128s
#pragma unroll
    for (int t = 0; t < 8; ++t) {
      v16h bV = *(const v16h*)&sVt[t * 16 + ln16][16 * hi];
      acc[t] = __builtin_amdgcn_wmma_f32_16x16x32_f16(false, aP, false, bV,
                                                      (short)0, acc[t], false, false);
    }
  }

  // ---- epilogue: mLSTM normalizer + per-head group norm over DH
  float inv[8];
#pragma unroll
  for (int vv = 0; vv < 8; ++vv) {
    const int rowl = wr0 + vv + 8 * hi;
    float mld = sCsq[rowl] + Mi[vv];               // exact row max of log_D
    float nrm = fmaxf(fabsf(sumAcc[vv]), expf(-mld)) + 1e-6f;
    inv[vv] = 1.f / nrm;
  }
  float msum[8], ssum[8];
#pragma unroll
  for (int vv = 0; vv < 8; ++vv) {
    float ls = 0.f, lq = 0.f;
#pragma unroll
    for (int t = 0; t < 8; ++t) {
      float hv = acc[t][vv] * inv[vv];
      ls += hv;
      lq += hv * hv;
    }
    msum[vv] = hsum16(ls) * (1.f / DH_);
    ssum[vv] = hsum16(lq) * (1.f / DH_);
  }
#pragma unroll
  for (int t = 0; t < 8; ++t) {
    const float w = nw[h * DH_ + t * 16 + ln16];
#pragma unroll
    for (int vv = 0; vv < 8; ++vv) {
      float mean = msum[vv];
      float var  = ssum[vv] - mean * mean;
      float hv   = acc[t][vv] * inv[vv];
      float o    = (hv - mean) * rsqrtf(var + 1e-5f) * w;
      const int rowl = wr0 + vv + 8 * hi;
      out[((size_t)b * S_ + (r0 + rowl)) * E_ + h * DH_ + t * 16 + ln16] = o;
    }
  }
}

// ---------------------------------------------------------------------------
extern "C" void kernel_launch(void* const* d_in, const int* in_sizes, int n_in,
                              void* d_out, int out_size, void* d_ws, size_t ws_size,
                              hipStream_t stream) {
  const float* q   = (const float*)d_in[0];
  const float* k   = (const float*)d_in[1];
  const float* v   = (const float*)d_in[2];
  const float* igw = (const float*)d_in[3];
  const float* igb = (const float*)d_in[4];
  const float* fgw = (const float*)d_in[5];
  const float* fgb = (const float*)d_in[6];
  const float* nw  = (const float*)d_in[7];
  float* out = (float*)d_out;

  const size_t N = (size_t)B_ * NH_ * S_;
  float* ipre = (float*)d_ws;     // B*NH*S
  float* fpre = ipre + N;
  float* cs   = fpre + N;
  float* Mv   = cs + N;           // prefix max of g
  float* Gv   = Mv + N;           // g = i_pre - cs

  mlstm_gates_kernel<<<B_ * S_, 128, 0, stream>>>(q, k, v, igw, igb, fgw, fgb,
                                                  ipre, fpre);
  mlstm_scan_kernel<<<B_ * NH_, 32, 0, stream>>>(fpre, ipre, cs, Mv, Gv);
  mlstm_attn_kernel<<<B_ * NH_ * (S_ / ROWS), NTHR, 0, stream>>>(q, k, v, Gv, Mv,
                                                                 cs, nw, out);
}
